// SelfAttentionHead_30494267802055
// MI455X (gfx1250) — matrix operand order repaired and checked by reading, not compiled, not run on hardware
//
#include <hip/hip_runtime.h>

typedef __attribute__((ext_vector_type(16))) _Float16 v16h;
typedef __attribute__((ext_vector_type(8)))  _Float16 v8h;
typedef __attribute__((ext_vector_type(8)))  float    v8f;
typedef __attribute__((ext_vector_type(4)))  float    v4f;

#define BB 256
#define TT 256
#define CC 384
#define HS 64

// workspace layout
static constexpr size_t WPACK_BYTES = 3ull * 12 * 4 * 32 * 16 * sizeof(_Float16); // 147456
static constexpr size_t QK_BYTES    = (size_t)BB * TT * HS * sizeof(_Float16);    // 8 MiB

// CDNA5 async global->LDS copy (per-lane 16B), tracked by ASYNCcnt
static __device__ __forceinline__ void async_ld_b128(unsigned lds_off, const void* gaddr) {
    asm volatile("global_load_async_to_lds_b128 %0, %1, off"
                 :: "v"(lds_off), "v"(gaddr) : "memory");
}
static __device__ __forceinline__ void wait_async0() {
    asm volatile("s_wait_asynccnt 0x0" ::: "memory");
}

// ---------------------------------------------------------------------------
// Kernel 1: pack Wq/Wk/Wv (fp32 [C][HS]) into the exact 32x16 f16 B-operand
// layout: [w][kstep 12][ntile 4][lane 32][16 halves]
// ---------------------------------------------------------------------------
__global__ void __launch_bounds__(256)
pack_w_kernel(const float* __restrict__ Wq, const float* __restrict__ Wk,
              const float* __restrict__ Wv, _Float16* __restrict__ wpack) {
    int idx = blockIdx.x * blockDim.x + threadIdx.x;      // [3][12][4][32]
    if (idx >= 3 * 12 * 4 * 32) return;
    int lane = idx % 32;
    int t1   = idx / 32;
    int nt   = t1 % 4;
    int t2   = t1 / 4;
    int ks   = t2 % 12;
    int w    = t2 / 12;
    const float* W = (w == 0) ? Wq : (w == 1) ? Wk : Wv;
    int k0 = ks * 32 + 8 * (lane >> 4);
    int n  = nt * 16 + (lane & 15);
    _Float16* dst = wpack + (size_t)idx * 16;
#pragma unroll
    for (int h = 0; h < 16; ++h)
        dst[h] = (_Float16)W[(size_t)(k0 + h + (h >= 8 ? 8 : 0)) * HS + n];
}

// ---------------------------------------------------------------------------
// Kernel 2: QKV projection. One wave per 16-row strip of x[B*T, C].
// Weights async-staged into LDS in 3 phases of 48KB (4 k-steps each), so
// B operands come from ds_load_b128 (~30cy) instead of L2 (~300cy).
// q,k stored f16 row-major [B*T][HS]; v stored f16 transposed [B][HS][T].
// ---------------------------------------------------------------------------
__global__ void __launch_bounds__(256)
qkv_kernel(const float* __restrict__ x, const _Float16* __restrict__ wpack,
           _Float16* __restrict__ qh, _Float16* __restrict__ kh,
           _Float16* __restrict__ vt) {
    // [w 3][ks_local 4][nt 4][lane 32][16 halves] = 49152 B
    __shared__ __align__(16) _Float16 smem[3 * 4 * 4 * 32 * 16];

    const int wave  = threadIdx.x >> 5;
    const int lane  = threadIdx.x & 31;
    const int l16   = lane & 15;
    const int hi    = lane >> 4;                 // 0 or 1
    const int mtile = blockIdx.x * 8 + wave;     // 0..4095
    const int row   = mtile * 16 + l16;          // A-operand row for this lane
    const float* xrow = x + (size_t)row * CC;
    const unsigned lds_base = (unsigned)(uintptr_t)smem;
    const char* wpack_bytes = (const char*)wpack;

    v8f accq[4] = {}, acck[4] = {}, accv[4] = {};

    // prologue: raw A loads for global k-step 0
    v4f ra0 = *(const v4f*)(xrow + hi * 8 +  0);
    v4f ra1 = *(const v4f*)(xrow + hi * 8 +  4);
    v4f ra2 = *(const v4f*)(xrow + hi * 8 + 16);
    v4f ra3 = *(const v4f*)(xrow + hi * 8 + 20);

    for (int phase = 0; phase < 3; ++phase) {
        __syncthreads();                          // prev phase consumers done
        // ---- async-stage 49152 B of packed weights: this phase's 4 k-steps
        // global layout: per w (stride 49152B), phase chunk of 16384B
#pragma unroll
        for (int j = 0; j < 12; ++j) {
            const unsigned linear = (unsigned)threadIdx.x * 16u + (unsigned)j * 4096u;
            const unsigned w      = linear >> 14;            // /16384
            const unsigned rem    = linear & 16383u;
            async_ld_b128(lds_base + linear,
                          wpack_bytes + (size_t)w * 49152 + (size_t)phase * 16384 + rem);
        }
        wait_async0();
        __syncthreads();

        for (int ksl = 0; ksl < 4; ++ksl) {
            const int g = phase * 4 + ksl;        // global k-step 0..11
            // ---- convert current A tile
            v16h a;
#pragma unroll
            for (int i = 0; i < 4; ++i) {
                a[i]      = (_Float16)ra0[i];
                a[4 + i]  = (_Float16)ra1[i];
                a[8 + i]  = (_Float16)ra2[i];
                a[12 + i] = (_Float16)ra3[i];
            }
            // ---- pipeline next A raw loads (global) under the WMMAs
            if (g < 11) {
                const float* xr = xrow + (g + 1) * 32 + hi * 8;
                ra0 = *(const v4f*)(xr +  0);
                ra1 = *(const v4f*)(xr +  4);
                ra2 = *(const v4f*)(xr + 16);
                ra3 = *(const v4f*)(xr + 20);
            }
            // ---- B tiles from LDS, 12 WMMAs
#pragma unroll
            for (int nt = 0; nt < 4; ++nt) {
                const v16h bq = *(const v16h*)(smem + (size_t)(((0 * 4 + ksl) * 4 + nt) * 32 + lane) * 16);
                const v16h bk = *(const v16h*)(smem + (size_t)(((1 * 4 + ksl) * 4 + nt) * 32 + lane) * 16);
                const v16h bv = *(const v16h*)(smem + (size_t)(((2 * 4 + ksl) * 4 + nt) * 32 + lane) * 16);
                accq[nt] = __builtin_amdgcn_wmma_f32_16x16x32_f16(false, a, false, bq, (short)0, accq[nt], false, false);
                acck[nt] = __builtin_amdgcn_wmma_f32_16x16x32_f16(false, a, false, bk, (short)0, acck[nt], false, false);
                accv[nt] = __builtin_amdgcn_wmma_f32_16x16x32_f16(false, a, false, bv, (short)0, accv[nt], false, false);
            }
        }
    }

    // ---- store: C layout lane holds (M = r + 8*hi, N = l16) per N-tile
#pragma unroll
    for (int nt = 0; nt < 4; ++nt) {
        const int col = nt * 16 + l16;
#pragma unroll
        for (int r = 0; r < 8; ++r) {
            const int grow = mtile * 16 + r + 8 * hi;        // global row in [0, B*T)
            qh[(size_t)grow * HS + col] = (_Float16)accq[nt][r];
            kh[(size_t)grow * HS + col] = (_Float16)acck[nt][r];
            const int b = grow >> 8;                          // / TT
            const int t = grow & 255;                         // % TT
            vt[((size_t)b * HS + col) * TT + t] = (_Float16)accv[nt][r];
        }
    }
}

// ---------------------------------------------------------------------------
// Kernel 3: causal flash attention. One wave per 16-query tile, keys in
// chunks of 32. K tiles double-pumped (next chunk's loads issued right after
// the score WMMAs, hidden under softmax); V tiles hidden the same way.
// ---------------------------------------------------------------------------
__global__ void __launch_bounds__(128)
attn_kernel(const _Float16* __restrict__ qh, const _Float16* __restrict__ kh,
            const _Float16* __restrict__ vt, float* __restrict__ out) {
    __shared__ __align__(32) _Float16 pbuf[4][16 * 32];      // 1 KB per wave

    const int wave = threadIdx.x >> 5;
    const int lane = threadIdx.x & 31;
    const int l16  = lane & 15;
    const int hi   = lane >> 4;
    const int qt   = blockIdx.x * 4 + wave;                  // 0..4095
    const int b    = qt >> 4;                                // 16 tiles per batch
    const int qb   = (qt & 15) * 16;                         // query-tile base row

    const _Float16* qbase = qh + (size_t)b * TT * HS;
    const _Float16* kbase = kh + (size_t)b * TT * HS;
    const _Float16* vbase = vt + (size_t)b * HS * TT;
    _Float16* lds = pbuf[wave];

    // ---- preload Q as two A tiles (HS = 64 -> 2 k-steps of 32)
    v16h qa[2];
#pragma unroll
    for (int ks = 0; ks < 2; ++ks) {
        const _Float16* p = qbase + (size_t)(qb + l16) * HS + ks * 32 + hi * 16;
        v8h lo = *(const v8h*)(p);
        v8h hix = *(const v8h*)(p + 8);
#pragma unroll
        for (int i = 0; i < 8; ++i) { qa[ks][i] = lo[i]; qa[ks][8 + i] = hix[i]; }
    }

    // softmax in base-2 domain: exp(x*s) == exp2(x * s*log2(e))
    const float sc2 = 0.125f * 1.4426950408889634f;
    float m[8], l[8];
    v8f acc[4] = {};
#pragma unroll
    for (int r = 0; r < 8; ++r) { m[r] = -1.0e30f; l[r] = 0.0f; }

    // ---- preload first K chunk (4 B-tiles)
    v16h kb[4];
#pragma unroll
    for (int sh = 0; sh < 2; ++sh)
#pragma unroll
        for (int ks = 0; ks < 2; ++ks)
            kb[sh * 2 + ks] = *(const v16h*)(kbase + (size_t)(sh * 16 + l16) * HS + ks * 32 + hi * 16);

    for (int s0 = 0; s0 <= qb; s0 += 32) {                   // wave-uniform trip count
        // ---- issue V B-tile loads; consumed only after softmax
        v16h vb[4];
#pragma unroll
        for (int nt = 0; nt < 4; ++nt)
            vb[nt] = *(const v16h*)(vbase + (size_t)(nt * 16 + l16) * TT + s0 + hi * 16);

        // ---- S = Q K^T for 32 keys (2 s-halves x 2 k-steps)
        v8f s[2] = {};
#pragma unroll
        for (int sh = 0; sh < 2; ++sh)
#pragma unroll
            for (int ks = 0; ks < 2; ++ks)
                s[sh] = __builtin_amdgcn_wmma_f32_16x16x32_f16(false, qa[ks], false, kb[sh * 2 + ks], (short)0, s[sh], false, false);

        // ---- refill kb for next chunk NOW; latency hides under softmax
        if (s0 + 32 <= qb) {
            const int sn = s0 + 32;
#pragma unroll
            for (int sh = 0; sh < 2; ++sh)
#pragma unroll
                for (int ks = 0; ks < 2; ++ks)
                    kb[sh * 2 + ks] = *(const v16h*)(kbase + (size_t)(sn + sh * 16 + l16) * HS + ks * 32 + hi * 16);
        }

        // ---- scale + causal mask + online softmax (rows live across 16 lanes)
#pragma unroll
        for (int r = 0; r < 8; ++r) {
            const int grow = qb + r + 8 * hi;
            float v0 = s[0][r] * sc2;
            float v1 = s[1][r] * sc2;
            if (s0 + l16 > grow)      v0 = -1.0e30f;
            if (s0 + 16 + l16 > grow) v1 = -1.0e30f;
            float rmax = fmaxf(v0, v1);
#pragma unroll
            for (int off = 1; off < 16; off <<= 1)
                rmax = fmaxf(rmax, __shfl_xor(rmax, off, 32));
            const float mnew  = fmaxf(m[r], rmax);
            const float alpha = exp2f(m[r] - mnew);
            const float p0 = exp2f(v0 - mnew);
            const float p1 = exp2f(v1 - mnew);
            float rsum = p0 + p1;
#pragma unroll
            for (int off = 1; off < 16; off <<= 1)
                rsum += __shfl_xor(rsum, off, 32);
            l[r] = l[r] * alpha + rsum;
            m[r] = mnew;
#pragma unroll
            for (int nt = 0; nt < 4; ++nt) acc[nt][r] *= alpha;
            // stash P into LDS in row-major [16][32]
            lds[(r + 8 * hi) * 32 + l16]      = (_Float16)p0;
            lds[(r + 8 * hi) * 32 + 16 + l16] = (_Float16)p1;
        }

        // ---- reload P in A layout (DS ops are in-order within a wave)
        v16h pa;
        {
            const _Float16* pr = lds + l16 * 32 + hi * 16;
            v8h lo = *(const v8h*)(pr);
            v8h hix = *(const v8h*)(pr + 8);
#pragma unroll
            for (int i = 0; i < 8; ++i) { pa[i] = lo[i]; pa[8 + i] = hix[i]; }
        }

        // ---- acc += P x V  (V tiles already resident in registers)
#pragma unroll
        for (int nt = 0; nt < 4; ++nt)
            acc[nt] = __builtin_amdgcn_wmma_f32_16x16x32_f16(false, pa, false, vb[nt], (short)0, acc[nt], false, false);
    }

    // ---- normalize and store fp32 output [B][T][HS]
#pragma unroll
    for (int nt = 0; nt < 4; ++nt) {
#pragma unroll
        for (int r = 0; r < 8; ++r) {
            const float o = acc[nt][r] / l[r];
            out[((size_t)b * TT + qb + r + 8 * hi) * HS + nt * 16 + l16] = o;
        }
    }
}

// ---------------------------------------------------------------------------
extern "C" void kernel_launch(void* const* d_in, const int* in_sizes, int n_in,
                              void* d_out, int out_size, void* d_ws, size_t ws_size,
                              hipStream_t stream) {
    (void)in_sizes; (void)n_in; (void)out_size; (void)ws_size;
    const float* x  = (const float*)d_in[0];
    const float* Wq = (const float*)d_in[1];
    const float* Wk = (const float*)d_in[2];
    const float* Wv = (const float*)d_in[3];
    float* out = (float*)d_out;

    char* ws = (char*)d_ws;
    _Float16* wpack = (_Float16*)ws;
    _Float16* qh    = (_Float16*)(ws + WPACK_BYTES);
    _Float16* kh    = (_Float16*)(ws + WPACK_BYTES + QK_BYTES);
    _Float16* vt    = (_Float16*)(ws + WPACK_BYTES + 2 * QK_BYTES);

    pack_w_kernel<<<(3 * 12 * 4 * 32 + 255) / 256, 256, 0, stream>>>(Wq, Wk, Wv, wpack);
    qkv_kernel<<<(BB * TT / 16) / 8, 256, 0, stream>>>(x, wpack, qh, kh, vt);
    attn_kernel<<<(BB * TT / 16) / 4, 128, 0, stream>>>(qh, kh, vt, out);
}
